// SNNR_Model_1614907703544
// MI455X (gfx1250) — matrix operand order repaired and checked by reading, not compiled, not run on hardware
//
#include <hip/hip_runtime.h>
#include <hip/hip_bf16.h>

#define DEVINL __device__ __forceinline__

typedef _Float16 h8v  __attribute__((ext_vector_type(8)));
typedef _Float16 h16v __attribute__((ext_vector_type(16)));
typedef float    f8v  __attribute__((ext_vector_type(8)));
typedef float    f2v  __attribute__((ext_vector_type(2)));

#if defined(__has_builtin)
#  if __has_builtin(__builtin_amdgcn_wmma_f32_16x16x4_f32)
#    define HAVE_WMMA_F32 1
#  else
#    define HAVE_WMMA_F32 0
#  endif
#else
#  define HAVE_WMMA_F32 0
#endif

namespace cfg {
constexpr int SEQ = 32, B = 64, DIN = 37632, H1 = 768, NC = 101;
constexpr int H2P = 112;                  // H2=100 padded to 7 col-tiles
constexpr int KC  = 128;                  // zc K padded for K%32==0
constexpr int NWG = 24;                   // persistent scan workgroups
constexpr int KSEG = 64, KLEN = DIN / KSEG;   // 588 per K-segment (147 K4-iters)
constexpr int NGRP = 24;                  // 48 n-tiles / 2-wide n blocking
constexpr float DTAU = 0.1f;              // DT*TAU
constexpr float LI_A = 0.1f;              // DT*LI_TAU_MEM
constexpr float LI_S = 0.2f;              // DT*LI_TAU_SYN
constexpr float VTH  = 0.3f;

// workspace offsets (bytes, all 256B aligned)
constexpr size_t OFF_INF   = 0;                                  // 64*768 f32
constexpr size_t OFF_VF    = OFF_INF   + (size_t)B*H1*4;
constexpr size_t OFF_IF    = OFF_VF    + (size_t)B*H1*4;
constexpr size_t OFF_VC    = OFF_IF    + (size_t)B*H1*4;
constexpr size_t OFF_IC    = OFF_VC    + (size_t)B*H2P*4;
constexpr size_t OFF_VO    = OFF_IC    + (size_t)B*H2P*4;
constexpr size_t OFF_IO    = OFF_VO    + (size_t)B*H2P*4;
constexpr size_t OFF_ZF    = OFF_IO    + (size_t)B*H2P*4;        // 3 x 64*768 f16
constexpr size_t OFF_ZC    = OFF_ZF    + (size_t)3*B*H1*2;       // 3 x 64*128 f16
constexpr size_t OFF_WREC  = OFF_ZC    + (size_t)3*B*KC*2;       // 768*768 f16
constexpr size_t OFF_WCIN  = OFF_WREC  + (size_t)H1*H1*2;        // 112*768 f16
constexpr size_t OFF_WCREC = OFF_WCIN  + (size_t)H2P*H1*2;       // 112*128 f16
constexpr size_t OFF_WOUT  = OFF_WCREC + (size_t)H2P*KC*2;       // 112*128 f16
constexpr size_t OFF_BAR   = OFF_WOUT  + (size_t)H2P*KC*2;       // 2 ints
constexpr size_t OFF_PART  = OFF_BAR   + 256;  // 1536 waves * 8 slots * 256 f32
}

DEVINL f8v zero8() { f8v z = {0.f,0.f,0.f,0.f,0.f,0.f,0.f,0.f}; return z; }

// A fragment (16x32 f16): lane L -> row L%16, chunks at k+(L/16)*8 and +16
DEVINL h16v ld_afrag(const _Float16* p) {
  h16v a;
  *reinterpret_cast<h8v*>(&a)       = *reinterpret_cast<const h8v*>(p);
  *(reinterpret_cast<h8v*>(&a) + 1) = *reinterpret_cast<const h8v*>(p + 16);
  return a;
}
// B fragment (32x16 f16): lane L -> col L%16, chunks at k+(L/16)*8 and +16
DEVINL h16v ld_bfrag(const _Float16* p) {
  h16v b;
  *reinterpret_cast<h8v*>(&b)       = *reinterpret_cast<const h8v*>(p);
  *(reinterpret_cast<h8v*>(&b) + 1) = *reinterpret_cast<const h8v*>(p + 16);
  return b;
}

DEVINL f8v wmma_f16(h16v a, h16v b, f8v c) {
  return __builtin_amdgcn_wmma_f32_16x16x32_f16(false, a, false, b, (short)0, c,
                                                false, false);
}

// C += A[m0:m0+16, :K] * B^T (B row-major W[n][k]); dual accumulation chains
// to break the WMMA->WMMA D->C dependency stall. Requires K % 64 == 0.
DEVINL f8v tile_gemm(const _Float16* A, int ldA, const _Float16* Bm, int ldB,
                     int m0, int n0, int K, int lane, f8v acc) {
  const _Float16* ap = A  + (size_t)(m0 + (lane & 15)) * ldA + ((lane >> 4) << 3);
  const _Float16* bp = Bm + (size_t)(n0 + (lane & 15)) * ldB + ((lane >> 4) << 3);
  f8v acc1 = zero8();
#pragma unroll 2
  for (int kb = 0; kb < K; kb += 64) {
    acc  = wmma_f16(ld_afrag(ap + kb),      ld_bfrag(bp + kb),      acc);
    acc1 = wmma_f16(ld_afrag(ap + kb + 32), ld_bfrag(bp + kb + 32), acc1);
  }
  return acc + acc1;
}

DEVINL void global_barrier(int* bar) {
  __threadfence();            // release this WG's spike/state stores (device scope)
  __syncthreads();
  if (threadIdx.x == 0) {
    int* cnt = bar;
    int* gen = bar + 1;
    int g = __hip_atomic_load(gen, __ATOMIC_RELAXED, __HIP_MEMORY_SCOPE_AGENT);
    int a = __hip_atomic_fetch_add(cnt, 1, __ATOMIC_ACQ_REL, __HIP_MEMORY_SCOPE_AGENT);
    if (a == cfg::NWG - 1) {
      __hip_atomic_store(cnt, 0, __ATOMIC_RELAXED, __HIP_MEMORY_SCOPE_AGENT);
      __hip_atomic_fetch_add(gen, 1, __ATOMIC_RELEASE, __HIP_MEMORY_SCOPE_AGENT);
    } else {
      while (__hip_atomic_load(gen, __ATOMIC_ACQUIRE, __HIP_MEMORY_SCOPE_AGENT) == g)
        __builtin_amdgcn_s_sleep(2);
    }
  }
  __syncthreads();
  __threadfence();            // acquire other WGs' stores
}

// ---------------- init: zero state, convert/pad weights to f16 ----------------
__global__ void __launch_bounds__(256) k_init(
    const float* __restrict__ Wf_rec, const float* __restrict__ Wc_in,
    const float* __restrict__ Wc_rec, const float* __restrict__ W_out,
    float* vf, float* if_, float* vc, float* ic, float* vo, float* io,
    _Float16* zf, _Float16* zc,
    _Float16* Wrec, _Float16* Wcin, _Float16* Wcrec, _Float16* Wout, int* bar) {
  const long i0 = (long)blockIdx.x * 256 + threadIdx.x;
  const long st = (long)gridDim.x * 256;
  for (long i = i0; i < 64L * 768; i += st) { vf[i] = 0.f; if_[i] = 0.f; }
  for (long i = i0; i < 64L * 112; i += st) { vc[i] = 0.f; ic[i] = 0.f; vo[i] = 0.f; io[i] = 0.f; }
  for (long i = i0; i < 3L * 64 * 768; i += st) zf[i] = (_Float16)0.f;
  for (long i = i0; i < 3L * 64 * 128; i += st) zc[i] = (_Float16)0.f;
  for (long i = i0; i < 768L * 768; i += st) Wrec[i] = (_Float16)Wf_rec[i];
  for (long i = i0; i < 112L * 768; i += st) {
    long n = i / 768, k = i % 768;
    Wcin[i] = (_Float16)(n < 100 ? Wc_in[n * 768 + k] : 0.f);
  }
  for (long i = i0; i < 112L * 128; i += st) {
    long n = i / 128, k = i % 128;
    Wcrec[i] = (_Float16)((n < 100 && k < 100) ? Wc_rec[n * 100 + k] : 0.f);
    Wout[i]  = (_Float16)((n < 101 && k < 100) ? W_out[n * 100 + k] : 0.f);
  }
  if (i0 == 0) { bar[0] = 0; bar[1] = 0; }
}

// ---- big fp32 GEMM in_f = x @ Wf_in^T: 4m x 2n register-blocked fp32 WMMA ----
// wave = (ks, ng): all 4 batch m-tiles x 2 output n-tiles, K-segment ks.
// 8 independent accumulator chains; 6 b64 loads per 8 WMMAs; Wf_in read once.
__global__ void __launch_bounds__(256) k_gemm_inf(const float* __restrict__ x,
                                                  const float* __restrict__ W,
                                                  float* __restrict__ part) {
  const int lane = threadIdx.x & 31;
  const int wid  = blockIdx.x * 8 + (threadIdx.x >> 5);  // 0..1535
  const int ng = wid % cfg::NGRP;                        // n-group (2 n-tiles)
  const int ks = wid / cfg::NGRP;                        // 0..63
  const long k0 = (long)ks * cfg::KLEN;
  f8v acc[8];
#pragma unroll
  for (int i = 0; i < 8; ++i) acc[i] = zero8();
#if HAVE_WMMA_F32
  // fp32 A 16x4: lane L -> row L%16, VGPR0/1 = K (L/16)*2, +1 -> b64 loads
  const float* pa = x + (size_t)(lane & 15) * cfg::DIN + k0 + ((lane >> 4) << 1);
  const float* pb = W + (size_t)(ng * 32 + (lane & 15)) * cfg::DIN + k0 + ((lane >> 4) << 1);
  for (int k = 0; k < cfg::KLEN; k += 4) {
    f2v b0 = *(const f2v*)(pb + k);
    f2v b1 = *(const f2v*)(pb + k + (size_t)16 * cfg::DIN);
    f2v a0 = *(const f2v*)(pa + k);
    f2v a1 = *(const f2v*)(pa + k + (size_t)16 * cfg::DIN);
    f2v a2 = *(const f2v*)(pa + k + (size_t)32 * cfg::DIN);
    f2v a3 = *(const f2v*)(pa + k + (size_t)48 * cfg::DIN);
    acc[0] = __builtin_amdgcn_wmma_f32_16x16x4_f32(false, a0, false, b0, (short)0, acc[0], false, false);
    acc[1] = __builtin_amdgcn_wmma_f32_16x16x4_f32(false, a0, false, b1, (short)0, acc[1], false, false);
    acc[2] = __builtin_amdgcn_wmma_f32_16x16x4_f32(false, a1, false, b0, (short)0, acc[2], false, false);
    acc[3] = __builtin_amdgcn_wmma_f32_16x16x4_f32(false, a1, false, b1, (short)0, acc[3], false, false);
    acc[4] = __builtin_amdgcn_wmma_f32_16x16x4_f32(false, a2, false, b0, (short)0, acc[4], false, false);
    acc[5] = __builtin_amdgcn_wmma_f32_16x16x4_f32(false, a2, false, b1, (short)0, acc[5], false, false);
    acc[6] = __builtin_amdgcn_wmma_f32_16x16x4_f32(false, a3, false, b0, (short)0, acc[6], false, false);
    acc[7] = __builtin_amdgcn_wmma_f32_16x16x4_f32(false, a3, false, b1, (short)0, acc[7], false, false);
  }
#else
  for (int k = 0; k < cfg::KLEN; ++k) {
    for (int slot = 0; slot < 8; ++slot) {
      const int m = slot >> 1, nl = slot & 1;
      const int row = m * 16 + ((lane >> 4) << 3);
      const int col = (ng * 2 + nl) * 16 + (lane & 15);
      float wv = W[(size_t)col * cfg::DIN + k0 + k];
      for (int r = 0; r < 8; ++r)
        acc[slot][r] += x[(size_t)(row + r) * cfg::DIN + k0 + k] * wv;
    }
  }
#endif
  // partial slot = m*2 + nl
  float* dst = part + ((size_t)wid * 8) * 256;
#pragma unroll
  for (int slot = 0; slot < 8; ++slot)
#pragma unroll
    for (int r = 0; r < 8; ++r) dst[slot * 256 + r * 32 + lane] = acc[slot][r];
}

// deterministic fixed-order K-segment reduction into in_f
__global__ void __launch_bounds__(256) k_reduce_inf(const float* __restrict__ part,
                                                    float* __restrict__ inf_) {
  const int tile = blockIdx.x;   // 0..191 (= m + 4*n)
  const int j = threadIdx.x;
  const int m = tile & 3, n = tile >> 2;
  const int ng = n >> 1, nl = n & 1;
  const int slot = m * 2 + nl;
  float s = 0.f;
  for (int ks = 0; ks < cfg::KSEG; ++ks)
    s += part[(((size_t)(ks * cfg::NGRP + ng)) * 8 + slot) * 256 + j];
  const int lane = j & 31, r = j >> 5;
  const int b = m * 16 + ((lane >> 4) << 3) + r;
  const int c = n * 16 + (lane & 15);
  inf_[b * cfg::H1 + c] = s;
}

// --------------------- persistent 32-step scan kernel ------------------------
__global__ void __launch_bounds__(256) k_scan(
    const float* __restrict__ inf_,
    float* vf, float* if_, float* vc, float* ic, float* vo, float* io,
    _Float16* zf, _Float16* zc,
    const _Float16* __restrict__ Wrec, const _Float16* __restrict__ Wcin,
    const _Float16* __restrict__ Wcrec, const _Float16* __restrict__ Wout,
    int* bar, float* __restrict__ out) {
  __shared__ __align__(16) _Float16 ldsW[32 * 768];   // this WG's 32 rows of Wrec
  const int g = blockIdx.x, tid = threadIdx.x;
  const int wave = tid >> 5, lane = tid & 31;

  {  // stage Wrec slice once, reused for all 32 steps (L2 -> LDS)
    const uint4* src = (const uint4*)(Wrec + (size_t)g * 32 * 768);
    uint4* dst = (uint4*)ldsW;
#pragma unroll
    for (int i = 0; i < 12; ++i) dst[tid + 256 * i] = src[tid + 256 * i];
  }
  __syncthreads();

  const int fm = wave >> 1;                 // f-tile: wave -> (m, n_local)
  const int fn_local = wave & 1;
  const int fcol0 = g * 32 + fn_local * 16;
  const int ct0 = g;                         // owned c/o tile(s): ct = m + 4*n
  const int ct1 = (g < 4) ? 24 + g : -1;

  for (int t = 0; t < cfg::SEQ; ++t) {
    _Float16* zf_cur        = zf + (size_t)(t % 3) * (64 * 768);
    const _Float16* zf_prev = zf + (size_t)((t + 2) % 3) * (64 * 768);
    _Float16* zc_cur        = zc + (size_t)(t % 3) * (64 * 128);
    const _Float16* zc_prev = zc + (size_t)((t + 2) % 3) * (64 * 128);

    // ---- phase A: elementwise spike/voltage updates + output write ----
    for (int i = tid; i < 64 * 32; i += 256) {
      int b = i >> 5, c = g * 32 + (i & 31);
      int idx = b * 768 + c;
      float vd = vf[idx] * (1.f - cfg::DTAU) + cfg::DTAU * if_[idx];
      float z = vd > cfg::VTH ? 1.f : 0.f;
      vf[idx] = vd * (1.f - z);
      zf_cur[idx] = (_Float16)z;
    }
    for (int s = 0; s < 2; ++s) {
      int ct = s ? ct1 : ct0;
      if (ct < 0) break;
      int m = ct & 3, n = ct >> 2;
      int b = m * 16 + (tid >> 4), c = n * 16 + (tid & 15);
      int idx = b * cfg::H2P + c;
      float vd = vc[idx] * (1.f - cfg::DTAU) + cfg::DTAU * ic[idx];
      float z = vd > cfg::VTH ? 1.f : 0.f;
      vc[idx] = vd * (1.f - z);
      zc_cur[b * cfg::KC + c] = (_Float16)z;
      float vn = vo[idx] * (1.f - cfg::LI_A) + cfg::LI_A * io[idx];
      vo[idx] = vn;
      if (c < cfg::NC) out[(size_t)t * 64 * cfg::NC + b * cfg::NC + c] = vn;
    }

    global_barrier(bar);   // publish spikes; one barrier/step (triple-buffered z)

    // ---- phase C: WMMA GEMMs + synaptic-current updates (owner WG only) ----
    {  // layer f recurrent: i_f = 0.9*i_f + in_f + zf_prev @ Wf_rec^T  (B in LDS)
      f8v acc = zero8();
      acc = tile_gemm(zf_prev, 768, ldsW, 768, fm * 16, fn_local * 16, 768, lane, acc);
      const int b0 = fm * 16 + ((lane >> 4) << 3);
      const int col = fcol0 + (lane & 15);
#pragma unroll
      for (int r = 0; r < 8; ++r) {
        int idx = (b0 + r) * 768 + col;
        if_[idx] = if_[idx] * (1.f - cfg::DTAU) + inf_[idx] + acc[r];
      }
    }
    for (int s = 0; s < 2; ++s) {
      int ct = s ? ct1 : ct0;
      if (ct < 0) break;
      int m = ct & 3, n = ct >> 2;
      int w23 = s ? 2 : 0, w4 = s ? 3 : 1;
      if (wave == w23) {  // i_c = 0.9*i_c + zf_cur@Wc_in^T + zc_prev@Wc_rec^T
        f8v acc = zero8();
        acc = tile_gemm(zf_cur, 768, Wcin, 768, m * 16, n * 16, 768, lane, acc);
        acc = tile_gemm(zc_prev, 128, Wcrec, 128, m * 16, n * 16, 128, lane, acc);
        const int b0 = m * 16 + ((lane >> 4) << 3);
        const int col = n * 16 + (lane & 15);
#pragma unroll
        for (int r = 0; r < 8; ++r) {
          int idx = (b0 + r) * cfg::H2P + col;
          ic[idx] = ic[idx] * (1.f - cfg::DTAU) + acc[r];
        }
      }
      if (wave == w4) {   // i_o = 0.8*i_o + zc_cur @ W_out^T
        f8v acc = zero8();
        acc = tile_gemm(zc_cur, 128, Wout, 128, m * 16, n * 16, 128, lane, acc);
        const int b0 = m * 16 + ((lane >> 4) << 3);
        const int col = n * 16 + (lane & 15);
#pragma unroll
        for (int r = 0; r < 8; ++r) {
          int idx = (b0 + r) * cfg::H2P + col;
          io[idx] = io[idx] * (1.f - cfg::LI_S) + acc[r];
        }
      }
    }
    __syncthreads();  // C(t) state writes visible to this WG's A(t+1)
  }
}

extern "C" void kernel_launch(void* const* d_in, const int* in_sizes, int n_in,
                              void* d_out, int out_size, void* d_ws, size_t ws_size,
                              hipStream_t stream) {
  (void)in_sizes; (void)n_in; (void)out_size; (void)ws_size;
  const float* x      = (const float*)d_in[0];
  const float* Wf_in  = (const float*)d_in[1];
  const float* Wf_rec = (const float*)d_in[2];
  const float* Wc_in  = (const float*)d_in[3];
  const float* Wc_rec = (const float*)d_in[4];
  const float* W_out  = (const float*)d_in[5];

  char* ws = (char*)d_ws;
  float*     inf_  = (float*)(ws + cfg::OFF_INF);
  float*     vf    = (float*)(ws + cfg::OFF_VF);
  float*     if_   = (float*)(ws + cfg::OFF_IF);
  float*     vc    = (float*)(ws + cfg::OFF_VC);
  float*     ic    = (float*)(ws + cfg::OFF_IC);
  float*     vo    = (float*)(ws + cfg::OFF_VO);
  float*     io    = (float*)(ws + cfg::OFF_IO);
  _Float16*  zf    = (_Float16*)(ws + cfg::OFF_ZF);
  _Float16*  zc    = (_Float16*)(ws + cfg::OFF_ZC);
  _Float16*  Wrec  = (_Float16*)(ws + cfg::OFF_WREC);
  _Float16*  Wcin  = (_Float16*)(ws + cfg::OFF_WCIN);
  _Float16*  Wcrec = (_Float16*)(ws + cfg::OFF_WCREC);
  _Float16*  Wout  = (_Float16*)(ws + cfg::OFF_WOUT);
  int*       bar   = (int*)(ws + cfg::OFF_BAR);
  float*     part  = (float*)(ws + cfg::OFF_PART);

  k_init<<<1024, 256, 0, stream>>>(Wf_rec, Wc_in, Wc_rec, W_out,
                                   vf, if_, vc, ic, vo, io, zf, zc,
                                   Wrec, Wcin, Wcrec, Wout, bar);
  k_gemm_inf<<<192, 256, 0, stream>>>(x, Wf_in, part);   // 1536 waves, 4x2 tiles
  k_reduce_inf<<<192, 256, 0, stream>>>(part, inf_);
  k_scan<<<cfg::NWG, 256, 0, stream>>>(inf_, vf, if_, vc, ic, vo, io, zf, zc,
                                       Wrec, Wcin, Wcrec, Wout, bar,
                                       (float*)d_out);
}
